// Attn_40767829573965
// MI455X (gfx1250) — matrix operand order repaired and checked by reading, not compiled, not run on hardware
//
#include <hip/hip_runtime.h>

// ---------------- types ----------------
typedef __bf16 bf16_t;
typedef __attribute__((ext_vector_type(8)))  __bf16 bfx8;
typedef __attribute__((ext_vector_type(16))) __bf16 bfx16;
typedef __attribute__((ext_vector_type(8)))  float  fx8;
typedef __attribute__((ext_vector_type(4)))  float  fx4;
typedef __attribute__((ext_vector_type(4)))  int    i32x4;

#define EMB    1024
#define SEQ    2048
#define HEADS  16
#define HD     64
#define BATCH  2

// ---- optional CDNA5 async global->LDS path (guarded; falls back to plain loads) ----
#if defined(__has_builtin)
#if __has_builtin(__builtin_amdgcn_global_load_async_to_lds_b128) && \
    __has_builtin(__builtin_amdgcn_s_wait_asynccnt)
#define USE_ASYNC_LDS 1
#endif
#endif

// builtin signature (from hipcc diagnostic): (v4i AS1*, v4i AS3*, imm off, imm cpol)
#define ASYNC_GSRC(p) ((__attribute__((address_space(1))) i32x4*)(p))
#define ASYNC_LDST(p) ((__attribute__((address_space(3))) i32x4*)(p))

static __device__ __forceinline__ fx8 wmma_bf16(bfx16 a, bfx16 b, fx8 c) {
    return __builtin_amdgcn_wmma_f32_16x16x32_bf16(false, a, false, b,
                                                   (short)0, c, false, false);
}

// ---------------- f32 -> bf16 convert (8 elems / thread) ----------------
__global__ __launch_bounds__(256) void cvt_f32_bf16(const float* __restrict__ src,
                                                    bf16_t* __restrict__ dst, int n8) {
    int i = blockIdx.x * 256 + threadIdx.x;
    if (i < n8) {
        fx4 a = ((const fx4*)src)[2 * i];
        fx4 b = ((const fx4*)src)[2 * i + 1];
        bfx8 o;
        o[0]=(bf16_t)a[0]; o[1]=(bf16_t)a[1]; o[2]=(bf16_t)a[2]; o[3]=(bf16_t)a[3];
        o[4]=(bf16_t)b[0]; o[5]=(bf16_t)b[1]; o[6]=(bf16_t)b[2]; o[7]=(bf16_t)b[3];
        ((bfx8*)dst)[i] = o;
    }
}

// ---------------- fused QKV GEMM: y = x @ W^T + b, scattered to [B,H,N,D] ----------------
// M = 4096, K = 1024, N = 3072. One wave computes a 64x64 tile (4x4 WMMA tiles):
// 16 b128 loads feed 16 WMMAs per K-step (1.0 wmma/load vs 0.5 at 2x2).
__global__ __launch_bounds__(256) void qkv_gemm(const bf16_t* __restrict__ xb,
                                                const bf16_t* __restrict__ wb,
                                                const float*  __restrict__ bias,
                                                bf16_t* __restrict__ qkv) {
    const int K = EMB;
    const int N64 = (3 * EMB) / 64;            // 48
    int wid  = (blockIdx.x * 256 + threadIdx.x) >> 5;
    int lane = threadIdx.x & 31;
    int lr = lane & 15;                        // row/col within 16
    int lh = lane >> 4;                        // half select
    int m4 = wid / N64, n4 = wid % N64;
    int mbase = m4 * 64, nbase = n4 * 64;

    const bf16_t* ap = xb + (size_t)(mbase + lr) * K + lh * 16;
    const bf16_t* bp = wb + (size_t)(nbase + lr) * K + lh * 16;

    fx8 acc[4][4] = {};
    for (int k0 = 0; k0 < K; k0 += 32) {
        bfx16 a[4], b[4];
#pragma unroll
        for (int mt = 0; mt < 4; ++mt) {
            const bf16_t* p = ap + (size_t)(mt * 16) * K + k0;
            bfx8 c0 = *(const bfx8*)p;
            bfx8 c1 = *(const bfx8*)(p + 8);
            a[mt] = __builtin_shufflevector(c0, c1, 0,1,2,3,4,5,6,7,8,9,10,11,12,13,14,15);
        }
#pragma unroll
        for (int nt = 0; nt < 4; ++nt)
            b[nt] = *(const bfx16*)(bp + (size_t)(nt * 16) * K + k0);
#pragma unroll
        for (int mt = 0; mt < 4; ++mt)
#pragma unroll
            for (int nt = 0; nt < 4; ++nt)
                acc[mt][nt] = wmma_bf16(a[mt], b[nt], acc[mt][nt]);
    }

    // Epilogue: add bias, convert to bf16, scatter into q/k/v [B,H,N,D]
    const size_t PART = (size_t)BATCH * HEADS * SEQ * HD;
#pragma unroll
    for (int nt = 0; nt < 4; ++nt) {
        int f    = nbase + nt * 16 + lr;       // feature index in [0, 3072)
        float bv = bias[f];
        int part = f >> 10;                    // 0=q,1=k,2=v
        int rem  = f & 1023;
        int h    = rem >> 6;
        int d    = rem & 63;
        bf16_t* dbase = qkv + (size_t)part * PART;
#pragma unroll
        for (int mt = 0; mt < 4; ++mt) {
            int mrow0 = mbase + mt * 16 + lh * 8;   // C layout: VGPR r -> row r (+8 high half)
            for (int r = 0; r < 8; ++r) {
                int m  = mrow0 + r;
                int bb = m >> 11;              // m / SEQ
                int n  = m & 2047;             // m % SEQ
                dbase[(((size_t)bb * HEADS + h) * SEQ + n) * HD + d] =
                    (bf16_t)(acc[mt][nt][r] + bv);
            }
        }
    }
}

// ---------------- flash attention ----------------
// Block = 256 threads = 8 waves; each wave owns 16 query rows (128 rows/block).
// Key loop in tiles of 32; K/V double-buffered in LDS; V stored transposed [d][key].
// K tile copied via GLOBAL_LOAD_ASYNC_TO_LDS_B128 when available (ASYNCcnt-tracked).
__global__ __launch_bounds__(256) void attn_fwd(const bf16_t* __restrict__ qws,
                                                const bf16_t* __restrict__ kws,
                                                const bf16_t* __restrict__ vws,
                                                float* __restrict__ out) {
    __shared__ __align__(16) bf16_t Kt[2][32][64];   // [buf][key][d]
    __shared__ __align__(16) bf16_t Vt[2][64][32];   // [buf][d][key]  (transposed)
    __shared__ __align__(16) bf16_t Pt[8][16][48];   // per-wave P bounce, padded stride

    int bh   = blockIdx.x >> 4;                // batch*head in [0,32)
    int qblk = blockIdx.x & 15;
    int b = bh >> 4, h = bh & 15;
    const bf16_t* qp = qws + (size_t)bh * SEQ * HD;
    const bf16_t* kp = kws + (size_t)bh * SEQ * HD;
    const bf16_t* vp = vws + (size_t)bh * SEQ * HD;

    int tid  = threadIdx.x;
    int wave = tid >> 5, lane = tid & 31;
    int lr = lane & 15, lh = lane >> 4;
    int qrow0 = qblk * 128 + wave * 16;

    // Q fragments: 16 rows x 64 d, as two 16x32 A fragments
    bfx16 qa[2];
#pragma unroll
    for (int j = 0; j < 2; ++j) {
        const bf16_t* p = qp + (size_t)(qrow0 + lr) * HD + j * 32 + lh * 16;
        bfx8 c0 = *(const bfx8*)p;
        bfx8 c1 = *(const bfx8*)(p + 8);
        qa[j] = __builtin_shufflevector(c0, c1, 0,1,2,3,4,5,6,7,8,9,10,11,12,13,14,15);
    }

    fx8 o[4] = {};
    float mrow[8], lrow[8];
    for (int r = 0; r < 8; ++r) { mrow[r] = -1e30f; lrow[r] = 0.f; }

    // cooperative K/V tile loader: 256 threads x 16B covers one 32x64 bf16 tile
    auto load_tiles = [&](int kt, int buf) {
        int row = tid >> 3;                // key within tile 0..31
        int col = (tid & 7) * 8;           // d chunk
        const bf16_t* kg = kp + (size_t)(kt * 32 + row) * HD + col;
        const bf16_t* vg = vp + (size_t)(kt * 32 + row) * HD + col;
#if defined(USE_ASYNC_LDS)
        // async byte-copy straight into LDS, no VGPR round-trip (ASYNCcnt-tracked)
        __builtin_amdgcn_global_load_async_to_lds_b128(
            ASYNC_GSRC(kg), ASYNC_LDST(&Kt[buf][row][col]), 0, 0);
#else
        *(bfx8*)&Kt[buf][row][col] = *(const bfx8*)kg;
#endif
        bfx8 vv = *(const bfx8*)vg;
        for (int i = 0; i < 8; ++i) Vt[buf][col + i][row] = vv[i];   // transpose
        if (kt + 2 < SEQ / 32) {           // prefetch next-next tile into cache
            __builtin_prefetch(kg + 64 * HD, 0, 0);
            __builtin_prefetch(vg + 64 * HD, 0, 0);
        }
    };

    load_tiles(0, 0);
#if defined(USE_ASYNC_LDS)
    __builtin_amdgcn_s_wait_asynccnt(0);
#endif
    __syncthreads();

    const float scale = 0.03125f;          // 1/sqrt(EMB) = 1/32 (reference scaling)
    const int NKT = SEQ / 32;              // 64
    for (int kt = 0; kt < NKT; ++kt) {
        int cur = kt & 1;
        if (kt + 1 < NKT) load_tiles(kt + 1, cur ^ 1);

        // S = Q K^T : two 16x16 f32 tiles (keys t*16..t*16+15), contraction over d=64
        fx8 s[2];
#pragma unroll
        for (int t = 0; t < 2; ++t) {
            bfx16 kb0 = *(const bfx16*)&Kt[cur][t * 16 + lr][lh * 16];
            bfx16 kb1 = *(const bfx16*)&Kt[cur][t * 16 + lr][32 + lh * 16];
            fx8 z = {};
            z    = wmma_bf16(qa[0], kb0, z);
            s[t] = wmma_bf16(qa[1], kb1, z);
        }

        // online softmax: each lane owns 8 rows (row = lh*8 + r), cols across 16 lanes
        float p0[8], p1[8];
#pragma unroll
        for (int r = 0; r < 8; ++r) {
            float mt_ = fmaxf(s[0][r], s[1][r]) * scale;
            for (int off = 8; off >= 1; off >>= 1)
                mt_ = fmaxf(mt_, __shfl_xor(mt_, off, 32));
            float mn    = fmaxf(mrow[r], mt_);
            float alpha = __expf(mrow[r] - mn);
            mrow[r] = mn;
            float a0 = __expf(s[0][r] * scale - mn);
            float a1 = __expf(s[1][r] * scale - mn);
            p0[r] = a0; p1[r] = a1;
            float rs = a0 + a1;
            for (int off = 8; off >= 1; off >>= 1)
                rs += __shfl_xor(rs, off, 32);
            lrow[r] = lrow[r] * alpha + rs;
#pragma unroll
            for (int dt = 0; dt < 4; ++dt) o[dt][r] *= alpha;
        }

        // P (C-layout) -> LDS -> A-layout bf16 fragment (per-wave LDS is in-order)
#pragma unroll
        for (int r = 0; r < 8; ++r) {
            Pt[wave][lh * 8 + r][lr]      = (bf16_t)p0[r];
            Pt[wave][lh * 8 + r][16 + lr] = (bf16_t)p1[r];
        }
        bfx8 pc0 = *(const bfx8*)&Pt[wave][lr][lh * 16];
        bfx8 pc1 = *(const bfx8*)&Pt[wave][lr][lh * 16 + 8];
        bfx16 pa = __builtin_shufflevector(pc0, pc1, 0,1,2,3,4,5,6,7,8,9,10,11,12,13,14,15);

        // O += P V : 4 d-tiles, contraction over 32 keys
#pragma unroll
        for (int dt = 0; dt < 4; ++dt) {
            bfx16 vb = *(const bfx16*)&Vt[cur][dt * 16 + lr][lh * 16];
            o[dt] = wmma_bf16(pa, vb, o[dt]);
        }
#if defined(USE_ASYNC_LDS)
        __builtin_amdgcn_s_wait_asynccnt(0);
#endif
        __syncthreads();
    }

    // normalize and store f32 output [B, SEQ, EMB]
#pragma unroll
    for (int dt = 0; dt < 4; ++dt)
        for (int r = 0; r < 8; ++r) {
            int row = qrow0 + lh * 8 + r;
            out[((size_t)b * SEQ + row) * EMB + h * HD + dt * 16 + lr] =
                o[dt][r] / lrow[r];
        }
}

// ---------------- host launch ----------------
extern "C" void kernel_launch(void* const* d_in, const int* in_sizes, int n_in,
                              void* d_out, int out_size, void* d_ws, size_t ws_size,
                              hipStream_t stream) {
    const float* x    = (const float*)d_in[0];   // [2, 2048, 1024]
    const float* w    = (const float*)d_in[1];   // [3072, 1024]
    const float* bias = (const float*)d_in[2];   // [3072]
    float* out = (float*)d_out;

    char* ws = (char*)d_ws;
    bf16_t* xb  = (bf16_t*)ws;                                   // 8 MiB
    bf16_t* wb  = (bf16_t*)(ws + (size_t)8  * 1024 * 1024);      // 6 MiB
    bf16_t* qkv = (bf16_t*)(ws + (size_t)14 * 1024 * 1024);      // 24 MiB (q,k,v)
    bf16_t* q = qkv;
    bf16_t* k = qkv + (size_t)BATCH * HEADS * SEQ * HD;
    bf16_t* v = k   + (size_t)BATCH * HEADS * SEQ * HD;

    const int NX = BATCH * SEQ * EMB;       // 4,194,304
    const int NW = 3 * EMB * EMB;           // 3,145,728
    cvt_f32_bf16<<<NX / 8 / 256, 256, 0, stream>>>(x, xb, NX / 8);
    cvt_f32_bf16<<<NW / 8 / 256, 256, 0, stream>>>(w, wb, NW / 8);

    // (4096/64) * (3072/64) = 3072 waves -> 384 blocks of 256
    qkv_gemm<<<384, 256, 0, stream>>>(xb, wb, bias, qkv);

    // 32 batch-heads * 16 query-blocks
    attn_fwd<<<512, 256, 0, stream>>>(q, k, v, out);
}
